// TiMAEMambaEncoder_85263690760218
// MI455X (gfx1250) — compile-verified, matrix operand order checked by reading, not run on hardware
//
#include <hip/hip_runtime.h>
#include <hip/hip_bf16.h>
#include <stdint.h>

// ---------------- problem constants ----------------
#define D_MODEL    512
#define INPUT_DIM  64
#define D_INNER    1024
#define D_STATE    128
#define NHEADS     16
#define HEADDIM    64
#define D_CONV     4
#define CONV_DIM   1280      // D_INNER + 2*D_STATE
#define D_IN_PROJ  2320
#define BATCH      2
#define SEQ        1024
#define LTOT       1025      // SEQ + 1 (cls)
#define MROWS      (BATCH*LTOT)   // 2050
#define MPAD       2176      // 17 * 128   (GEMM M tile = 128)
#define NPAD1      2432      // 19 * 128   (GEMM N tile = 128)
#define KDIM1      512
#define KDIM2      1024
#define NDIM2      512
#define LEN_KEEP   256
#define EPSF       1e-5f

typedef __attribute__((ext_vector_type(16))) __bf16 v16bf;
typedef __attribute__((ext_vector_type(8)))  __bf16 v8bf;
typedef __attribute__((ext_vector_type(8)))  float  v8f;

#if __has_builtin(__builtin_amdgcn_s_wait_asynccnt)
#define USE_ASYNC_LDS 1
#else
#define USE_ASYNC_LDS 0
#endif

static inline int ceil_div(int a, int b) { return (a + b - 1) / b; }

// 16B global -> LDS copy. Async path uses the CDNA5 GLOBAL_LOAD_ASYNC_TO_LDS_B128
// (ASYNCcnt-tracked DMA, bypasses VGPRs); vdst carries the 32-bit LDS byte offset
// (low half of the generic pointer to a __shared__ object).
__device__ __forceinline__ void copy16_g2l(const __bf16* g, __bf16* s) {
#if USE_ASYNC_LDS
    asm volatile("global_load_async_to_lds_b128 %0, %1, off"
                 :: "v"((uint32_t)(uintptr_t)s), "v"((uint64_t)(uintptr_t)g)
                 : "memory");
#else
    *(v8bf*)s = *(const v8bf*)g;
#endif
}

// ---------------- weight packing (f32 -> bf16, zero-padded) ----------------
__global__ void k_pack_w1(const float* __restrict__ win, __bf16* __restrict__ winb) {
    int idx = blockIdx.x * blockDim.x + threadIdx.x;     // over NPAD1*KDIM1
    if (idx >= NPAD1 * KDIM1) return;
    int r = idx / KDIM1;
    float v = (r < D_IN_PROJ) ? win[idx] : 0.f;          // padded rows -> 0
    winb[idx] = (__bf16)v;
}

__global__ void k_pack_w2(const float* __restrict__ wout, __bf16* __restrict__ woutb) {
    int idx = blockIdx.x * blockDim.x + threadIdx.x;     // over NDIM2*KDIM2
    if (idx >= NDIM2 * KDIM2) return;
    woutb[idx] = (__bf16)wout[idx];
}

// ---------------- conv embedding + cls concat -> padded bf16 A matrix ----------------
__global__ void k_embed(const float* __restrict__ x, const float* __restrict__ w,
                        const float* __restrict__ bias, const float* __restrict__ cls,
                        __bf16* __restrict__ hb) {
    int idx = blockIdx.x * blockDim.x + threadIdx.x;     // over MPAD*D_MODEL
    if (idx >= MPAD * D_MODEL) return;
    int r = idx / D_MODEL, o = idx % D_MODEL;
    float acc = 0.f;
    if (r < MROWS) {
        int b = r / LTOT, l = r % LTOT;
        if (l == SEQ) {
            acc = cls[o];
        } else {
            acc = bias[o];
            #pragma unroll
            for (int k = 0; k < 3; ++k) {
                int ls = l + k - 1;
                if (ls >= 0 && ls < SEQ) {
                    const float* xp = x + ((size_t)b * SEQ + ls) * INPUT_DIM;
                    const float* wp = w + (size_t)o * INPUT_DIM * 3 + k;
                    #pragma unroll
                    for (int i = 0; i < INPUT_DIM; ++i) acc += xp[i] * wp[i * 3];
                }
            }
        }
    }
    hb[idx] = (__bf16)acc;
}

// ---------------- WMMA bf16 GEMM: C[M,N] = A[M,K] * B[N,K]^T  (NT, f32 accum) ----------
// 256 threads = 8 waves; block tile 128(M) x 128(N); wave tile 32x64; K step 32.
// Double-buffered LDS; staging via GLOBAL_LOAD_ASYNC_TO_LDS (ASYNCcnt) when available.
__global__ __launch_bounds__(256)
void k_gemm_bf16(const __bf16* __restrict__ A, const __bf16* __restrict__ B,
                 float* __restrict__ C, int K, int ldc) {
    __shared__ __align__(16) __bf16 As[2][128][32];
    __shared__ __align__(16) __bf16 Bs[2][128][32];

    const int tid   = threadIdx.x;
    const int lane  = tid & 31;
    const int wave  = tid >> 5;
    const int waveM = wave >> 1;      // 0..3 -> 32 rows
    const int waveN = wave & 1;       // 0..1 -> 64 cols
    const int blockM = blockIdx.y * 128;
    const int blockN = blockIdx.x * 128;
    const int m  = lane & 15;         // fragment row/col within 16
    const int kh = lane >> 4;         // K-half select (ISA 16-bit layout)

    // staging coords: each thread owns 16B chunks {tid, tid+256} of each 128x32 tile
    const int r0 = tid >> 2;                 // rows 0..63
    const int r1 = r0 + 64;                  // rows 64..127
    const int c0 = (tid & 3) << 3;           // col offset 0/8/16/24

    v8f acc[2][4] = {};

    auto stage = [&](int kt, int buf) {
        copy16_g2l(A + (size_t)(blockM + r0) * K + kt + c0, &As[buf][r0][c0]);
        copy16_g2l(A + (size_t)(blockM + r1) * K + kt + c0, &As[buf][r1][c0]);
        copy16_g2l(B + (size_t)(blockN + r0) * K + kt + c0, &Bs[buf][r0][c0]);
        copy16_g2l(B + (size_t)(blockN + r1) * K + kt + c0, &Bs[buf][r1][c0]);
    };

    stage(0, 0);   // prologue: tile 0 in flight into buffer 0

    for (int kt = 0; kt < K; kt += 32) {
        const int buf = (kt >> 5) & 1;
#if USE_ASYNC_LDS
        __builtin_amdgcn_s_wait_asynccnt(0);   // my share of buffer `buf` has landed
#endif
        __syncthreads();                       // everyone's share landed; prev reads done
        if (kt + 32 < K) stage(kt + 32, buf ^ 1);

        // fragments per ISA 7.12.2: lane = row (lane&15), K-half (lane>>4):
        // VGPR0-3 <- k = kh*8 + 0..7 ; VGPR4-7 <- k = 16 + kh*8 + 0..7
        v16bf afr[2], bfr[4];
        #pragma unroll
        for (int i = 0; i < 2; ++i) {
            const __bf16* p = &As[buf][waveM * 32 + i * 16 + m][kh * 8];
            v8bf lo = *(const v8bf*)p;
            v8bf hi = *(const v8bf*)(p + 16);
            afr[i] = __builtin_shufflevector(lo, hi, 0,1,2,3,4,5,6,7,8,9,10,11,12,13,14,15);
        }
        #pragma unroll
        for (int j = 0; j < 4; ++j) {
            const __bf16* p = &Bs[buf][waveN * 64 + j * 16 + m][kh * 8];
            v8bf lo = *(const v8bf*)p;
            v8bf hi = *(const v8bf*)(p + 16);
            bfr[j] = __builtin_shufflevector(lo, hi, 0,1,2,3,4,5,6,7,8,9,10,11,12,13,14,15);
        }
        #pragma unroll
        for (int i = 0; i < 2; ++i)
            #pragma unroll
            for (int j = 0; j < 4; ++j)
                acc[i][j] = __builtin_amdgcn_wmma_f32_16x16x32_bf16(
                    false, afr[i], false, bfr[j], (short)0, acc[i][j], false, false);
    }

    // store D: VGPR v -> row base + 8*kh + v ; col = base + (lane&15)
    #pragma unroll
    for (int i = 0; i < 2; ++i) {
        #pragma unroll
        for (int j = 0; j < 4; ++j) {
            int rbase = blockM + waveM * 32 + i * 16 + kh * 8;
            int col   = blockN + waveN * 64 + j * 16 + m;
            #pragma unroll
            for (int v = 0; v < 8; ++v)
                C[(size_t)(rbase + v) * ldc + col] = acc[i][j][v];
        }
    }
}

// ---------------- dt = softplus(raw + dt_bias), dA = exp(dt * -exp(A_log)) ------------
__global__ void k_dt(const float* __restrict__ zx, const float* __restrict__ dt_bias,
                     const float* __restrict__ A_log,
                     float* __restrict__ dt_out, float* __restrict__ dA_out) {
    int idx = blockIdx.x * blockDim.x + threadIdx.x;     // MROWS*NHEADS
    if (idx >= MROWS * NHEADS) return;
    int r = idx / NHEADS, h = idx % NHEADS;
    float v = zx[(size_t)r * NPAD1 + (D_INNER + CONV_DIM) + h] + dt_bias[h];
    float dt = (v > 20.f) ? v : log1pf(expf(v));
    float Ah = -expf(A_log[h]);
    dt_out[idx] = dt;
    dA_out[idx] = expf(dt * Ah);
}

// ---------------- causal depthwise conv1d (k=4) + bias + silu ----------------
__global__ void k_dwconv(const float* __restrict__ zx, const float* __restrict__ w,
                         const float* __restrict__ bias, float* __restrict__ xbc) {
    int idx = blockIdx.x * blockDim.x + threadIdx.x;     // MROWS*CONV_DIM
    if (idx >= MROWS * CONV_DIM) return;
    int c = idx % CONV_DIM;
    int r = idx / CONV_DIM;
    int b = r / LTOT, l = r % LTOT;
    float acc = bias[c];
    #pragma unroll
    for (int k = 0; k < D_CONV; ++k) {
        int ls = l - (D_CONV - 1) + k;
        if (ls >= 0)
            acc += zx[((size_t)b * LTOT + ls) * NPAD1 + D_INNER + c] * w[c * D_CONV + k];
    }
    xbc[idx] = acc / (1.f + expf(-acc));                 // silu
}

// ---------------- selective-state scan: one block per (b,h), state in regs ------------
__global__ __launch_bounds__(256)
void k_scan(const float* __restrict__ xbc, const float* __restrict__ dt,
            const float* __restrict__ dA, float* __restrict__ y) {
    const int b = blockIdx.x >> 4;
    const int h = blockIdx.x & 15;
    const int tid = threadIdx.x;
    const int p  = tid >> 2;       // 0..63
    const int nc = tid & 3;        // 0..3
    const int n0 = nc * 32;

    float st[32];
    #pragma unroll
    for (int i = 0; i < 32; ++i) st[i] = 0.f;

    __shared__ float sB[D_STATE], sC[D_STATE], sx[HEADDIM];

    for (int l = 0; l < LTOT; ++l) {
        size_t row = (size_t)b * LTOT + l;
        const float* xr = xbc + row * CONV_DIM;
        if (tid < 128)      sB[tid]       = xr[D_INNER + tid];
        else                sC[tid - 128] = xr[D_INNER + D_STATE + (tid - 128)];
        if (tid < HEADDIM)  sx[tid]       = xr[h * HEADDIM + tid];
        __syncthreads();

        float dtv = dt[row * NHEADS + h];
        float dav = dA[row * NHEADS + h];
        float xb  = dtv * sx[p];
        float part = 0.f;
        #pragma unroll
        for (int i = 0; i < 32; ++i) {
            st[i] = st[i] * dav + xb * sB[n0 + i];
            part += st[i] * sC[n0 + i];
        }
        part += __shfl_xor(part, 1);
        part += __shfl_xor(part, 2);
        if (nc == 0) y[(row * NHEADS + h) * HEADDIM + p] = part;
        __syncthreads();
    }
}

// ---------------- y = (scan + D*x) * silu(z); RMSNorm; -> padded bf16 ---------------
__global__ __launch_bounds__(256)
void k_gate_rms(const float* __restrict__ zx, const float* __restrict__ ysc,
                const float* __restrict__ xbc, const float* __restrict__ Dv,
                const float* __restrict__ rms_w, __bf16* __restrict__ yb) {
    int r = blockIdx.x;                                  // 0..MPAD-1
    if (r >= MROWS) {
        for (int i = threadIdx.x; i < D_INNER; i += 256)
            yb[(size_t)r * D_INNER + i] = (__bf16)0.f;
        return;
    }
    __shared__ float red[8];
    size_t row = r;
    float vals[4];
    float ss = 0.f;
    #pragma unroll
    for (int it = 0; it < 4; ++it) {
        int e = threadIdx.x + it * 256;                  // 0..1023
        int h = e >> 6;
        float yv = ysc[(row * NHEADS + h) * HEADDIM + (e & 63)]
                 + Dv[h] * xbc[row * CONV_DIM + e];
        float z = zx[row * NPAD1 + e];
        yv *= z / (1.f + expf(-z));
        vals[it] = yv;
        ss += yv * yv;
    }
    #pragma unroll
    for (int o = 16; o > 0; o >>= 1) ss += __shfl_xor(ss, o);
    if ((threadIdx.x & 31) == 0) red[threadIdx.x >> 5] = ss;
    __syncthreads();
    float tot = red[0]+red[1]+red[2]+red[3]+red[4]+red[5]+red[6]+red[7];
    float inv = rsqrtf(tot * (1.f / D_INNER) + EPSF);
    #pragma unroll
    for (int it = 0; it < 4; ++it) {
        int e = threadIdx.x + it * 256;
        yb[row * D_INNER + e] = (__bf16)(vals[it] * inv * rms_w[e]);
    }
}

// ---------------- final LayerNorm over D_MODEL ----------------
__global__ __launch_bounds__(256)
void k_ln(const float* __restrict__ o, const float* __restrict__ g,
          const float* __restrict__ be, float* __restrict__ res) {
    int r = blockIdx.x;                                  // 0..MROWS-1
    int tid = threadIdx.x;
    __shared__ float red[8];
    float v0 = o[(size_t)r * D_MODEL + tid];
    float v1 = o[(size_t)r * D_MODEL + 256 + tid];
    float s = v0 + v1;
    #pragma unroll
    for (int k = 16; k > 0; k >>= 1) s += __shfl_xor(s, k);
    if ((tid & 31) == 0) red[tid >> 5] = s;
    __syncthreads();
    float mu = (red[0]+red[1]+red[2]+red[3]+red[4]+red[5]+red[6]+red[7]) * (1.f / D_MODEL);
    __syncthreads();
    float d0 = v0 - mu, d1 = v1 - mu;
    float s2 = d0 * d0 + d1 * d1;
    #pragma unroll
    for (int k = 16; k > 0; k >>= 1) s2 += __shfl_xor(s2, k);
    if ((tid & 31) == 0) red[tid >> 5] = s2;
    __syncthreads();
    float var = (red[0]+red[1]+red[2]+red[3]+red[4]+red[5]+red[6]+red[7]) * (1.f / D_MODEL);
    float inv = rsqrtf(var + EPSF);
    res[(size_t)r * D_MODEL + tid]       = d0 * inv * g[tid]       + be[tid];
    res[(size_t)r * D_MODEL + 256 + tid] = d1 * inv * g[256 + tid] + be[256 + tid];
}

// ---------------- masking: noise, rank (argsort), gather ----------------
__global__ void k_noise(float* __restrict__ noise) {
    int idx = blockIdx.x * blockDim.x + threadIdx.x;
    if (idx >= BATCH * SEQ) return;
    uint32_t xv = (uint32_t)idx * 0x9E3779B1u + 0x85EBCA77u;
    xv ^= xv >> 16; xv *= 0x7feb352du; xv ^= xv >> 15; xv *= 0x846ca68bu; xv ^= xv >> 16;
    noise[idx] = (xv >> 8) * (1.0f / 16777216.0f);
}

__global__ __launch_bounds__(256)
void k_rank(const float* __restrict__ noise, int* __restrict__ ids_shuf,
            int* __restrict__ ids_rest) {
    int b = blockIdx.x;
    __shared__ float sn[SEQ];
    for (int i = threadIdx.x; i < SEQ; i += 256) sn[i] = noise[b * SEQ + i];
    __syncthreads();
    for (int i = threadIdx.x; i < SEQ; i += 256) {
        float v = sn[i];
        int rk = 0;
        for (int j = 0; j < SEQ; ++j) {
            float u = sn[j];
            rk += (u < v) || (u == v && j < i);
        }
        ids_shuf[b * SEQ + rk] = i;     // inverse of rank permutation
        ids_rest[b * SEQ + i]  = rk;    // argsort(argsort) == rank
    }
}

__global__ void k_gather(const float* __restrict__ ln, const int* __restrict__ ids_shuf,
                         float* __restrict__ xout) {
    int idx = blockIdx.x * blockDim.x + threadIdx.x;     // BATCH*257*D_MODEL
    if (idx >= BATCH * (LEN_KEEP + 1) * D_MODEL) return;
    int c = idx % D_MODEL;
    int t = idx / D_MODEL;
    int b = t / (LEN_KEEP + 1);
    int j = t % (LEN_KEEP + 1);
    int srcl = (j == LEN_KEEP) ? SEQ : ids_shuf[b * SEQ + j];
    xout[idx] = ln[((size_t)b * LTOT + srcl) * D_MODEL + c];
}

__global__ void k_maskids(const int* __restrict__ ids_shuf, const int* __restrict__ ids_rest,
                          float* __restrict__ mask, int* __restrict__ orest,
                          int* __restrict__ okeep) {
    int idx = blockIdx.x * blockDim.x + threadIdx.x;     // BATCH*SEQ
    if (idx >= BATCH * SEQ) return;
    int b = idx / SEQ, s = idx % SEQ;
    int rk = ids_rest[idx];
    mask[idx]  = (rk < LEN_KEEP) ? 0.f : 1.f;
    orest[idx] = rk;
    if (s < LEN_KEEP) okeep[b * LEN_KEEP + s] = ids_shuf[idx];
}

// ---------------- orchestration ----------------
extern "C" void kernel_launch(void* const* d_in, const int* in_sizes, int n_in,
                              void* d_out, int out_size, void* d_ws, size_t ws_size,
                              hipStream_t stream) {
    (void)in_sizes; (void)n_in; (void)out_size; (void)ws_size;

    const float* x          = (const float*)d_in[0];
    const float* conv_emb_w = (const float*)d_in[1];
    const float* conv_emb_b = (const float*)d_in[2];
    const float* cls_token  = (const float*)d_in[3];
    const float* in_proj_w  = (const float*)d_in[4];
    const float* conv1d_w   = (const float*)d_in[5];
    const float* conv1d_b   = (const float*)d_in[6];
    const float* dt_bias    = (const float*)d_in[7];
    const float* A_log      = (const float*)d_in[8];
    const float* Dvec       = (const float*)d_in[9];
    const float* rms_w      = (const float*)d_in[10];
    const float* out_proj_w = (const float*)d_in[11];
    const float* ln_g       = (const float*)d_in[12];
    const float* ln_b       = (const float*)d_in[13];

    // workspace carve-up (256B aligned)
    char* w = (char*)d_ws;
    size_t off = 0;
    auto carve = [&](size_t bytes) {
        void* p = w + off;
        off = (off + bytes + 255) & ~(size_t)255;
        return p;
    };
    __bf16* hB    = (__bf16*)carve((size_t)MPAD * KDIM1 * 2);
    __bf16* WinB  = (__bf16*)carve((size_t)NPAD1 * KDIM1 * 2);
    __bf16* WoutB = (__bf16*)carve((size_t)NDIM2 * KDIM2 * 2);
    float*  ZX    = (float*) carve((size_t)MPAD * NPAD1 * 4);
    float*  XBC   = (float*) carve((size_t)MROWS * CONV_DIM * 4);
    float*  DT    = (float*) carve((size_t)MROWS * NHEADS * 4);
    float*  DA    = (float*) carve((size_t)MROWS * NHEADS * 4);
    float*  Y     = (float*) carve((size_t)MROWS * D_INNER * 4);
    __bf16* YB    = (__bf16*)carve((size_t)MPAD * KDIM2 * 2);
    float*  OUT   = (float*) carve((size_t)MPAD * NDIM2 * 4);
    float*  LNOUT = (float*) carve((size_t)MROWS * D_MODEL * 4);
    float*  NOISE = (float*) carve((size_t)BATCH * SEQ * 4);
    int*    ISHUF = (int*)   carve((size_t)BATCH * SEQ * 4);
    int*    IREST = (int*)   carve((size_t)BATCH * SEQ * 4);

    // output layout: x_out | mask | ids_restore | ids_keep
    float* xout  = (float*)d_out;
    float* maskp = xout + (size_t)BATCH * (LEN_KEEP + 1) * D_MODEL;
    int*   orest = (int*)(maskp + (size_t)BATCH * SEQ);
    int*   okeep = orest + (size_t)BATCH * SEQ;

    const int T = 256;

    k_pack_w1<<<ceil_div(NPAD1 * KDIM1, T), T, 0, stream>>>(in_proj_w, WinB);
    k_pack_w2<<<ceil_div(NDIM2 * KDIM2, T), T, 0, stream>>>(out_proj_w, WoutB);
    k_embed  <<<ceil_div(MPAD * D_MODEL, T), T, 0, stream>>>(x, conv_emb_w, conv_emb_b,
                                                             cls_token, hB);
    // GEMM1: ZX[MPAD, NPAD1] = hB[MPAD,512] * WinB[NPAD1,512]^T
    {
        dim3 g(NPAD1 / 128, MPAD / 128);
        k_gemm_bf16<<<g, T, 0, stream>>>(hB, WinB, ZX, KDIM1, NPAD1);
    }
    k_dt     <<<ceil_div(MROWS * NHEADS, T), T, 0, stream>>>(ZX, dt_bias, A_log, DT, DA);
    k_dwconv <<<ceil_div(MROWS * CONV_DIM, T), T, 0, stream>>>(ZX, conv1d_w, conv1d_b, XBC);
    k_scan   <<<BATCH * NHEADS, T, 0, stream>>>(XBC, DT, DA, Y);
    k_gate_rms<<<MPAD, T, 0, stream>>>(ZX, Y, XBC, Dvec, rms_w, YB);
    // GEMM2: OUT[MPAD, 512] = YB[MPAD,1024] * WoutB[512,1024]^T
    {
        dim3 g(NDIM2 / 128, MPAD / 128);
        k_gemm_bf16<<<g, T, 0, stream>>>(YB, WoutB, OUT, KDIM2, NDIM2);
    }
    k_ln     <<<MROWS, T, 0, stream>>>(OUT, ln_g, ln_b, LNOUT);
    k_noise  <<<ceil_div(BATCH * SEQ, T), T, 0, stream>>>(NOISE);
    k_rank   <<<BATCH, T, 0, stream>>>(NOISE, ISHUF, IREST);
    k_gather <<<ceil_div(BATCH * (LEN_KEEP + 1) * D_MODEL, T), T, 0, stream>>>(LNOUT, ISHUF, xout);
    k_maskids<<<ceil_div(BATCH * SEQ, T), T, 0, stream>>>(ISHUF, IREST, maskp, orest, okeep);
}
